// TripletAttention_48146583388345
// MI455X (gfx1250) — matrix operand order repaired and checked by reading, not compiled, not run on hardware
//
#include <hip/hip_runtime.h>

#define B_ 2
#define N_ 256
#define E_ 256
#define H_ 768
#define SQS 772   // padded q-tile row stride in floats (772%64==4 -> conflict-free)

typedef __attribute__((ext_vector_type(16))) __bf16    v16bf;
typedef __attribute__((ext_vector_type(16))) _Float16  v16h;
typedef __attribute__((ext_vector_type(8)))  _Float16  v8h;
typedef __attribute__((ext_vector_type(8)))  float     v8f;

// ---------------------------------------------------------------------------
// Prep 1: x (fp32, row-major [B*N][E]) -> bf16 row-major.
// ---------------------------------------------------------------------------
__global__ __launch_bounds__(256)
void cvt_x_kernel(const float* __restrict__ src, __bf16* __restrict__ dst)
{
    const int idx = (blockIdx.x * 256 + threadIdx.x) * 4;
    const float4 f = *(const float4*)(src + idx);
    __align__(8) __bf16 o[4] = { (__bf16)f.x, (__bf16)f.y, (__bf16)f.z, (__bf16)f.w };
    *(uint2*)(dst + idx) = *(const uint2*)o;
}

// ---------------------------------------------------------------------------
// Prep 2: W (fp32 [E][H]) -> bf16 transposed [H][E]  (wt[h*E+e] = W[e*H+h]).
// ---------------------------------------------------------------------------
__global__ __launch_bounds__(256)
void cvt_wt_kernel(const float* __restrict__ W, __bf16* __restrict__ wt)
{
    const int g  = blockIdx.x * 256 + threadIdx.x;   // over H*E/4
    const int h  = g >> 6;                           // E/4 == 64 e-groups per h
    const int e0 = (g & 63) * 4;
    __align__(8) __bf16 o[4];
    #pragma unroll
    for (int u = 0; u < 4; ++u)
        o[u] = (__bf16)W[(size_t)(e0 + u) * H_ + h];
    *(uint2*)(wt + (size_t)h * E_ + e0) = *(const uint2*)o;
}

// ---------------------------------------------------------------------------
// Kernel 1: fused Q/K/V projection, one wave per 16x16 tile, bf16 WMMA.
// A/B fragments are each two b128 loads (ISA 7.12.2 layouts, two 8-wide K runs
// per lane). Writes q, kv=k+v (fp32) and v transposed as f16 vt[b][h][j].
// ---------------------------------------------------------------------------
__global__ __launch_bounds__(32)
void proj_kernel(const __bf16* __restrict__ xbf,
                 const __bf16* __restrict__ wtq, const __bf16* __restrict__ wtk,
                 const __bf16* __restrict__ wtv,
                 const float* __restrict__ bq, const float* __restrict__ bk,
                 const float* __restrict__ bv,
                 float* __restrict__ qws, float* __restrict__ kvws,
                 _Float16* __restrict__ vt)
{
    const int lane = threadIdx.x;
    const int col  = lane & 15;   // A row / B,C column
    const int half = lane >> 4;
    const int h0 = blockIdx.x * 16;
    const int i0 = blockIdx.y * 16;
    const int b  = blockIdx.z;

    v8f cq = {}; v8f ck = {}; v8f cv = {};
    const __bf16* xrow = xbf + (size_t)(b * N_ + i0 + col) * E_;
    const __bf16* wq   = wtq + (size_t)(h0 + col) * E_;
    const __bf16* wk   = wtk + (size_t)(h0 + col) * E_;
    const __bf16* wv   = wtv + (size_t)(h0 + col) * E_;

    for (int k0 = 0; k0 < E_; k0 += 32) {
        v16bf a, fq, fk, fv;
        // A 16x32 bf16: VGPR v holds K = (v/4)*16 + half*8 + (v%4)*2 (+1)
        *(uint4*)&a       = *(const uint4*)(xrow + k0 + half * 8);
        *((uint4*)&a + 1) = *(const uint4*)(xrow + k0 + 16 + half * 8);
        // B 32x16 bf16: lane = column, VGPR v holds K = (v/4)*16 + half*8 + (v%4)*2 (+1)
        const int ko = k0 + half * 8;
        *(uint4*)&fq       = *(const uint4*)(wq + ko);
        *((uint4*)&fq + 1) = *(const uint4*)(wq + ko + 16);
        *(uint4*)&fk       = *(const uint4*)(wk + ko);
        *((uint4*)&fk + 1) = *(const uint4*)(wk + ko + 16);
        *(uint4*)&fv       = *(const uint4*)(wv + ko);
        *((uint4*)&fv + 1) = *(const uint4*)(wv + ko + 16);
        cq = __builtin_amdgcn_wmma_f32_16x16x32_bf16(false, a, false, fq, (short)0, cq, false, false);
        ck = __builtin_amdgcn_wmma_f32_16x16x32_bf16(false, a, false, fk, (short)0, ck, false, false);
        cv = __builtin_amdgcn_wmma_f32_16x16x32_bf16(false, a, false, fv, (short)0, cv, false, false);
    }

    const float bqv = bq[h0 + col];
    const float bkv = bk[h0 + col];
    const float bvv = bv[h0 + col];
    v8h pv;
    #pragma unroll
    for (int r = 0; r < 8; ++r) {
        // C: VGPR r -> M = r + 8*half, N = lane&15
        const int irow = i0 + r + half * 8;
        const size_t idx = (size_t)(b * N_ + irow) * H_ + h0 + col;
        const float vv = cv[r] + bvv;
        qws[idx]  = cq[r] + bqv;
        kvws[idx] = (ck[r] + bkv) + vv;
        pv[r] = (_Float16)vv;
    }
    // v transposed f16: vt[(b*H + h)*N + j], 8 consecutive j per lane (16B store)
    *(v8h*)(vt + ((size_t)(b * H_) + h0 + col) * N_ + i0 + half * 8) = pv;
}

// ---------------------------------------------------------------------------
// Kernel 2: energy + softmax + attn@v per (batch, 16-row i-tile). 8 waves.
// LDS: padded q-tile (async-staged, 49408B) + f16 energy (8192B);
// f16 attn buffer aliases the dead q-tile region after phase 2.
// ---------------------------------------------------------------------------
__global__ __launch_bounds__(256)
void attn_kernel(const float* __restrict__ qws, const float* __restrict__ kvws,
                 const _Float16* __restrict__ vt, const float* __restrict__ we,
                 const float* __restrict__ be_p, float* __restrict__ out)
{
    __shared__ __align__(16) unsigned char smem[16 * SQS * 4 + N_ * 16 * 2];
    float*     sq = (float*)smem;                      // sq[i*SQS + h]
    _Float16*  se = (_Float16*)(smem + 16 * SQS * 4);  // se[j*16 + i] (energy)
    _Float16*  af = (_Float16*)smem;                   // af[i*256 + j] (attn, aliases sq)

    const int t  = threadIdx.x;
    const int i0 = blockIdx.x * 16;
    const int b  = blockIdx.y;

    // Phase 1: async-stage q tile [16][H] into padded LDS rows.
    for (int f = t; f < 16 * (H_ / 4); f += 256) {     // 3072 16-byte chunks
        const int i = f / (H_ / 4);
        const int c = f - i * (H_ / 4);
        const float* gptr = qws + (size_t)(b * N_ + i0 + i) * H_ + c * 4;
        const unsigned lds_off = (unsigned)(uintptr_t)(sq + i * SQS + c * 4);
        asm volatile("global_load_async_to_lds_b128 %0, %1, off"
                     :: "v"(lds_off), "v"(gptr) : "memory");
    }
    asm volatile("s_wait_asynccnt 0x0" ::: "memory");
    __syncthreads();

    // Phase 2: energy[i][j] = sum_h we[h]*lrelu(q[i][h]+kv[j][h]) + be.
    {
        const float bev = be_p[0];
        const int i  = t & 15;
        const int jg = t >> 4;
        for (int j0 = 0; j0 < N_; j0 += 16) {
            const int j = j0 + jg;
            const float* kvrow = kvws + (size_t)(b * N_ + j) * H_;
            if (j0 + 16 < N_)
                __builtin_prefetch(kvrow + 16 * H_, 0, 0);
            const float* qrow = sq + i * SQS;
            float acc = 0.f;
            for (int h = 0; h < H_; h += 4) {
                const float4 q4  = *(const float4*)(qrow + h);
                const float4 kv4 = *(const float4*)(kvrow + h);
                const float4 w4  = *(const float4*)(we + h);
                const float t0 = q4.x + kv4.x;
                const float t1 = q4.y + kv4.y;
                const float t2 = q4.z + kv4.z;
                const float t3 = q4.w + kv4.w;
                acc = fmaf(w4.x, fmaxf(t0, 0.01f * t0), acc);
                acc = fmaf(w4.y, fmaxf(t1, 0.01f * t1), acc);
                acc = fmaf(w4.z, fmaxf(t2, 0.01f * t2), acc);
                acc = fmaf(w4.w, fmaxf(t3, 0.01f * t3), acc);
            }
            se[j * 16 + i] = (_Float16)(acc + bev);
        }
    }
    __syncthreads();   // after this, sq region is dead -> reusable as af

    // Phase 3: softmax over j per row; wave w owns rows 2w (lanes 0-15)
    // and 2w+1 (lanes 16-31); 16-lane reductions via __shfl_xor (wave32).
    {
        const int wave = t >> 5;
        const int lane = t & 31;
        const int row  = wave * 2 + (lane >> 4);
        const int jl   = lane & 15;
        float vals[16];
        float m = -1e30f;
        #pragma unroll
        for (int c = 0; c < 16; ++c) {
            const float x = (float)se[(jl + 16 * c) * 16 + row];
            vals[c] = x;
            m = fmaxf(m, x);
        }
        #pragma unroll
        for (int off = 1; off < 16; off <<= 1)
            m = fmaxf(m, __shfl_xor(m, off, 16));
        float s = 0.f;
        #pragma unroll
        for (int c = 0; c < 16; ++c) {
            vals[c] = __expf(vals[c] - m);
            s += vals[c];
        }
        #pragma unroll
        for (int off = 1; off < 16; off <<= 1)
            s += __shfl_xor(s, off, 16);
        const float inv = 1.f / s;
        #pragma unroll
        for (int c = 0; c < 16; ++c)
            af[row * 256 + jl + 16 * c] = (_Float16)(vals[c] * inv);
    }
    __syncthreads();

    // Phase 4: out[16][H] = attn[16][N] @ v[N][H] via f16 WMMA.
    // A from LDS (2 x ds b128), B from vt (2 x global b128) per fragment.
    {
        const int wave  = t >> 5;
        const int lane  = t & 31;
        const int colc  = lane & 15;
        const int halfc = lane >> 4;
        for (int ht = wave; ht < H_ / 16; ht += 8) {
            const int h0 = ht * 16;
            v8f acc = {};
            const _Float16* vrow = vt + ((size_t)(b * H_) + h0 + colc) * N_;
            for (int jb = 0; jb < N_; jb += 32) {
                v16h afrag, bfrag;
                const _Float16* pa = af + colc * 256 + jb + halfc * 8;
                *(uint4*)&afrag       = *(const uint4*)(pa);
                *((uint4*)&afrag + 1) = *(const uint4*)(pa + 16);
                const _Float16* pb = vrow + jb + halfc * 8;
                *(uint4*)&bfrag       = *(const uint4*)(pb);
                *((uint4*)&bfrag + 1) = *(const uint4*)(pb + 16);
                acc = __builtin_amdgcn_wmma_f32_16x16x32_f16(false, afrag, false, bfrag,
                                                             (short)0, acc, false, false);
            }
            #pragma unroll
            for (int r = 0; r < 8; ++r) {
                const int irow = i0 + r + halfc * 8;
                out[(size_t)(b * N_ + irow) * H_ + h0 + colc] = acc[r];
            }
        }
    }
}

// ---------------------------------------------------------------------------
extern "C" void kernel_launch(void* const* d_in, const int* in_sizes, int n_in,
                              void* d_out, int out_size, void* d_ws, size_t ws_size,
                              hipStream_t stream)
{
    const float* x  = (const float*)d_in[0];
    const float* Wq = (const float*)d_in[1];
    const float* bq = (const float*)d_in[2];
    const float* Wk = (const float*)d_in[3];
    const float* bk = (const float*)d_in[4];
    const float* Wv = (const float*)d_in[5];
    const float* bv = (const float*)d_in[6];
    const float* we = (const float*)d_in[7];
    const float* be = (const float*)d_in[8];
    float* out = (float*)d_out;

    const size_t plane = (size_t)B_ * N_ * H_;   // 393216
    char* w = (char*)d_ws;
    float*    qws  = (float*)w;     w += plane * 4;
    float*    kvws = (float*)w;     w += plane * 4;
    _Float16* vt   = (_Float16*)w;  w += plane * 2;
    __bf16*   xbf  = (__bf16*)w;    w += (size_t)B_ * N_ * E_ * 2;
    __bf16*   wtq  = (__bf16*)w;    w += (size_t)E_ * H_ * 2;
    __bf16*   wtk  = (__bf16*)w;    w += (size_t)E_ * H_ * 2;
    __bf16*   wtv  = (__bf16*)w;

    cvt_x_kernel <<<(B_ * N_ * E_) / (256 * 4), 256, 0, stream>>>(x, xbf);
    cvt_wt_kernel<<<(E_ * H_) / (256 * 4), 256, 0, stream>>>(Wq, wtq);
    cvt_wt_kernel<<<(E_ * H_) / (256 * 4), 256, 0, stream>>>(Wk, wtk);
    cvt_wt_kernel<<<(E_ * H_) / (256 * 4), 256, 0, stream>>>(Wv, wtv);

    dim3 gA(H_ / 16, N_ / 16, B_);
    proj_kernel<<<gA, 32, 0, stream>>>(xbf, wtq, wtk, wtv, bq, bk, bv, qws, kvws, vt);

    dim3 gB(N_ / 16, B_);
    attn_kernel<<<gB, 256, 0, stream>>>(qws, kvws, vt, we, be, out);
}
